// Encoder_74689481278119
// MI455X (gfx1250) — matrix operand order repaired and checked by reading, not compiled, not run on hardware
//
#include <hip/hip_runtime.h>

#define VN 20000
#define GN 16000
#define CN 8
#define MN 6
#define EN 64

typedef __attribute__((ext_vector_type(16))) __bf16          v16bf;
typedef __attribute__((ext_vector_type(8)))  float           v8f;
typedef __attribute__((ext_vector_type(8)))  unsigned short  u16x8;
typedef __attribute__((ext_vector_type(2)))  unsigned int    u32x2;

// CDNA5 has V_FMA_MIXLO_BF16 / V_FMA_MIXHI_BF16 (ISA 15.10 opcodes 62/63):
// one instruction = fma in f32 + convert-to-bf16 into half of a VGPR.
#define USE_MIX_CVT 1

__device__ __forceinline__ unsigned short f2bf_sw(float f) {
  unsigned x = __float_as_uint(f);
  x += 0x7fffu + ((x >> 16) & 1u);          // round-to-nearest-even
  return (unsigned short)(x >> 16);
}

#if USE_MIX_CVT
// bf16(x)
__device__ __forceinline__ unsigned short f2bf(float x) {
  unsigned r;
  asm("v_fma_mixlo_bf16 %0, %1, 1.0, 0" : "=v"(r) : "v"(x));
  return (unsigned short)r;
}
// bf16(x + b)  (bias-add fused into the convert)
__device__ __forceinline__ unsigned short fab2bf(float x, float b) {
  unsigned r;
  asm("v_fma_mixlo_bf16 %0, %1, 1.0, %2" : "=v"(r) : "v"(x), "v"(b));
  return (unsigned short)r;
}
// bf16(x * s)  (scale fused into the convert)
__device__ __forceinline__ unsigned short fms2bf(float x, float s) {
  unsigned r;
  asm("v_fma_mixlo_bf16 %0, %1, %2, 0" : "=v"(r) : "v"(x), "v"(s));
  return (unsigned short)r;
}
// {bf16(b), bf16(a)} packed into one dword (a -> [15:0], b -> [31:16])
__device__ __forceinline__ unsigned pk2(float a, float b) {
  unsigned r;
  asm("v_fma_mixlo_bf16 %0, %1, 1.0, 0\n\t"
      "v_fma_mixhi_bf16 %0, %2, 1.0, 0"
      : "=&v"(r) : "v"(a), "v"(b));
  return r;
}
#else
__device__ __forceinline__ unsigned short f2bf(float x) { return f2bf_sw(x); }
__device__ __forceinline__ unsigned short fab2bf(float x, float b) { return f2bf_sw(x + b); }
__device__ __forceinline__ unsigned short fms2bf(float x, float s) { return f2bf_sw(x * s); }
__device__ __forceinline__ unsigned pk2(float a, float b) {
  return (unsigned)f2bf_sw(a) | ((unsigned)f2bf_sw(b) << 16);
}
#endif

// A-fragment: two 16-byte chunks at p and p+32B (16 ushorts apart)
__device__ __forceinline__ v16bf frag_a(const unsigned short* p) {
  union { v16bf v; u16x8 h[2]; } u;
  u.h[0] = *(const u16x8*)(p);
  u.h[1] = *(const u16x8*)(p + 16);
  return u.v;
}
// B-fragment: 16 contiguous bf16 per lane (pre-swizzled in ws)
__device__ __forceinline__ v16bf frag_b(const unsigned short* p) {
  union { v16bf v; u16x8 h[2]; } u;
  u.h[0] = *(const u16x8*)(p);
  u.h[1] = *(const u16x8*)(p + 8);
  return u.v;
}

#define WMMA_BF16(A, B, Cc) \
  __builtin_amdgcn_wmma_f32_16x16x32_bf16(false, (A), false, (B), (short)0, (Cc), false, false)

// ---------------- weight pre-swizzle into B-fragment layout ----------------
// frag element (kt, nt, lane, e) = W[kt*32 + 16*(e/8) + 8*(lane/16) + e%8][nt*16 + lane%16]
#define NW_FRAG  (2  * 4 * 512)   // W_neg   64x64
#define NVC_FRAG (12 * 4 * 512)   // vc_W*  384x64
#define NCC_FRAG (16 * 4 * 512)   // cc_W*  512x64
#define NFRAG_TOT (NW_FRAG + 2 * NVC_FRAG + 2 * NCC_FRAG)   // 118784

__global__ void prep_weights_kernel(const float* __restrict__ Wneg,
                                    const float* __restrict__ vcW1,
                                    const float* __restrict__ vcW2,
                                    const float* __restrict__ ccW1,
                                    const float* __restrict__ ccW2,
                                    unsigned short* __restrict__ out) {
  int i = blockIdx.x * blockDim.x + threadIdx.x;
  const float* W;
  unsigned short* dst;
  int idx;
  if (i < NW_FRAG)                       { W = Wneg; idx = i;                         dst = out; }
  else if (i < NW_FRAG + NVC_FRAG)       { W = vcW1; idx = i - NW_FRAG;               dst = out + NW_FRAG; }
  else if (i < NW_FRAG + 2*NVC_FRAG)     { W = vcW2; idx = i - NW_FRAG - NVC_FRAG;    dst = out + NW_FRAG + NVC_FRAG; }
  else if (i < NW_FRAG + 2*NVC_FRAG + NCC_FRAG)
                                         { W = ccW1; idx = i - NW_FRAG - 2*NVC_FRAG;  dst = out + NW_FRAG + 2*NVC_FRAG; }
  else if (i < NFRAG_TOT)                { W = ccW2; idx = i - NW_FRAG - 2*NVC_FRAG - NCC_FRAG;
                                                                                      dst = out + NW_FRAG + 2*NVC_FRAG + NCC_FRAG; }
  else return;
  int e    = idx & 15;
  int lane = (idx >> 4) & 31;
  int nt   = (idx >> 9) & 3;
  int kt   = idx >> 11;
  int K = kt * 32 + ((e >> 3) << 4) + ((lane >> 4) << 3) + (e & 7);
  int N = (nt << 4) + (lane & 15);
  dst[idx] = f2bf(W[K * EN + N]);
}

// ---------------- initial variables: l2norm(emb_table) / l2norm(tseitin) ----
__global__ __launch_bounds__(32) void init_vars_kernel(const float* __restrict__ emb,
                                                       const float* __restrict__ tse,
                                                       float* __restrict__ V0) {
  int v = blockIdx.x, lane = threadIdx.x;
  const float* src = (v < GN) ? (emb + (long)v * EN) : tse;
  float x0 = src[lane], x1 = src[lane + 32];
  float s = x0 * x0 + x1 * x1;
  s += __shfl_xor(s, 1, 32);
  s += __shfl_xor(s, 2, 32);
  s += __shfl_xor(s, 4, 32);
  s += __shfl_xor(s, 8, 32);
  s += __shfl_xor(s, 16, 32);
  float inv = 1.f / fmaxf(sqrtf(s), 1e-12f);
  V0[(long)v * EN + lane]      = x0 * inv;
  V0[(long)v * EN + lane + 32] = x1 * inv;
}

// ---------------- fused message-passing iteration: 1 wave = 16 variables ----
// LDS layout (ushort units)
#define LB_OFF    0                      // 96 raw literal rows  x 72 bf16
#define LN_OFF    6912                   // 96 negated rows      x 72 bf16
#define FALSE_OFF 13824                  // 1 row (64 + pad)
#define Y_OFF     13904                  // 16 rows x 520 bf16 (c_emb tile, 16x512)
#define SMEM_USH  (Y_OFF + 16 * 520)     // 22224 ushorts (~44.4 KB)

__global__ __launch_bounds__(32, 4) void iter_kernel(
    const float* __restrict__ Vin, float* __restrict__ Vout,
    const int* __restrict__ clause_lits, const int* __restrict__ lit_neg,
    const int* __restrict__ lit_valid, const int* __restrict__ clause_valid,
    const float* __restrict__ false_emb, const float* __restrict__ b_neg,
    const float* __restrict__ vc_b1, const float* __restrict__ vc_b2,
    const float* __restrict__ cc_b1, const float* __restrict__ cc_b2,
    const unsigned short* __restrict__ frags) {
  __shared__ unsigned short smem[SMEM_USH];
  __shared__ unsigned rowSrc[16 * MN];   // bf16-row offsets into smem (select+permute)

  const unsigned short* fWneg = frags;
  const unsigned short* fvcW1 = frags + NW_FRAG;
  const unsigned short* fvcW2 = frags + NW_FRAG + NVC_FRAG;
  const unsigned short* fccW1 = frags + NW_FRAG + 2 * NVC_FRAG;
  const unsigned short* fccW2 = frags + NW_FRAG + 2 * NVC_FRAG + NCC_FRAG;

  const int lane = threadIdx.x;
  const int l16  = lane & 15;
  const int hlf  = lane >> 4;
  const int gvb  = blockIdx.x * 16;

  for (int e = lane; e < EN; e += 32) smem[FALSE_OFF + e] = f2bf(false_emb[e]);

  float bnegv[4], vb1[4], vb2[4], cb1[4], cb2[4];
#pragma unroll
  for (int nt = 0; nt < 4; ++nt) {
    int n = nt * 16 + l16;
    bnegv[nt] = b_neg[n];
    vb1[nt] = vc_b1[n]; vb2[nt] = vc_b2[n];
    cb1[nt] = cc_b1[n]; cb2[nt] = cc_b2[n];
  }

#pragma unroll 1
  for (int c = 0; c < CN; ++c) {
    // ---- gather 96 literal embeddings (bf16) into LDS; prefetch clause c+1 ----
#pragma unroll 1
    for (int t = 0; t < 3; ++t) {
      int s = lane * 3 + t;
      int v = s / 6, m = s - 6 * v;
      int base = ((gvb + v) * CN + c) * MN + m;
      int lit = clause_lits[base];
      if (c + 1 < CN) {  // warm WGP$ for next clause's gather (global_prefetch_b8)
        int nlit = clause_lits[base + MN];
        __builtin_prefetch(Vin + (long)nlit * EN, 0, 3);
      }
      const float4* src = (const float4*)(Vin + (long)lit * EN);
      unsigned* dst = (unsigned*)(smem + LB_OFF + s * 72);
#pragma unroll
      for (int q = 0; q < 16; ++q) {
        float4 f = src[q];
        u32x2 w;
        w.x = pk2(f.x, f.y);
        w.y = pk2(f.z, f.w);
        *(u32x2*)(dst + q * 2) = w;
      }
    }

    // ---- W_neg matmul over all 96 rows -> Ln (branchless writeback) ----
#pragma unroll 1
    for (int rt = 0; rt < 6; ++rt) {
      v8f acc[4] = {};
#pragma unroll
      for (int kt = 0; kt < 2; ++kt) {
        v16bf A = frag_a(smem + LB_OFF + (rt * 16 + l16) * 72 + kt * 32 + hlf * 8);
#pragma unroll
        for (int nt = 0; nt < 4; ++nt) {
          v16bf B = frag_b(fWneg + ((kt * 4 + nt) * 32 + lane) * 16);
          acc[nt] = WMMA_BF16(A, B, acc[nt]);
        }
      }
#pragma unroll
      for (int nt = 0; nt < 4; ++nt)
#pragma unroll
        for (int j = 0; j < 8; ++j) {
          int row = rt * 16 + j + hlf * 8;
          smem[LN_OFF + row * 72 + nt * 16 + l16] = fab2bf(acc[nt][j], bnegv[nt]);
        }
    }

    // ---- selection (neg/valid/false) + permutation folded into rowSrc ----
    if (lane < 16) {
      int v = lane;
      int base = ((gvb + v) * CN + c) * MN;
      int lits[MN], vld[MN], ngs[MN];
#pragma unroll
      for (int m = 0; m < MN; ++m) {
        lits[m] = clause_lits[base + m];
        vld[m]  = lit_valid[base + m];
        ngs[m]  = lit_neg[base + m];
      }
      int ind = 0;
#pragma unroll
      for (int m = MN - 1; m >= 0; --m)
        if (vld[m] > 0 && lits[m] == gvb + v) ind = m;   // first match wins
#pragma unroll
      for (int m = 0; m < MN; ++m) {
        int p = (m == ind) ? 0 : (m < ind ? m + 1 : m);
        unsigned rowoff = (unsigned)((v * MN + m) * 72);
        unsigned off = (vld[m] > 0)
                           ? (ngs[m] > 0 ? (LN_OFF + rowoff) : (LB_OFF + rowoff))
                           : (unsigned)FALSE_OFF;
        rowSrc[v * MN + p] = off;
      }
    }

    // ---- vc combine: 16x384 @ 384x64 (x2), sigmoid+add, l2norm, mask ----
    {
      v8f a1[4] = {}, a2[4] = {};
#pragma unroll 2
      for (int kt = 0; kt < 12; ++kt) {
        v16bf A = frag_a(smem + rowSrc[l16 * MN + (kt >> 1)] + (kt & 1) * 32 + hlf * 8);
#pragma unroll
        for (int nt = 0; nt < 4; ++nt) {
          v16bf B1 = frag_b(fvcW1 + ((kt * 4 + nt) * 32 + lane) * 16);
          a1[nt] = WMMA_BF16(A, B1, a1[nt]);
          v16bf B2 = frag_b(fvcW2 + ((kt * 4 + nt) * 32 + lane) * 16);
          a2[nt] = WMMA_BF16(A, B2, a2[nt]);
        }
      }
      float s[8] = {0.f, 0.f, 0.f, 0.f, 0.f, 0.f, 0.f, 0.f};
#pragma unroll
      for (int nt = 0; nt < 4; ++nt)
#pragma unroll
        for (int j = 0; j < 8; ++j) {
          float x1 = a1[nt][j] + vb1[nt];
          float val = 1.f / (1.f + __expf(-x1)) + a2[nt][j] + vb2[nt];
          a1[nt][j] = val;
          s[j] += val * val;
        }
#pragma unroll
      for (int j = 0; j < 8; ++j) {
        float t = s[j];
        t += __shfl_xor(t, 1, 32);
        t += __shfl_xor(t, 2, 32);
        t += __shfl_xor(t, 4, 32);
        t += __shfl_xor(t, 8, 32);
        s[j] = t;
      }
#pragma unroll
      for (int j = 0; j < 8; ++j) {
        int row = j + hlf * 8;
        int cv = clause_valid[(gvb + row) * CN + c];
        float inv = (cv > 0) ? (1.f / fmaxf(sqrtf(s[j]), 1e-12f)) : 0.f;
#pragma unroll
        for (int nt = 0; nt < 4; ++nt)
          smem[Y_OFF + row * 520 + c * 64 + nt * 16 + l16] = fms2bf(a1[nt][j], inv);
      }
    }
  }  // clause loop

  // ---- cc combine: 16x512 @ 512x64 (x2) -> new variables ----
  {
    v8f a1[4] = {}, a2[4] = {};
#pragma unroll 2
    for (int kt = 0; kt < 16; ++kt) {
      v16bf A = frag_a(smem + Y_OFF + l16 * 520 + kt * 32 + hlf * 8);
#pragma unroll
      for (int nt = 0; nt < 4; ++nt) {
        v16bf B1 = frag_b(fccW1 + ((kt * 4 + nt) * 32 + lane) * 16);
        a1[nt] = WMMA_BF16(A, B1, a1[nt]);
        v16bf B2 = frag_b(fccW2 + ((kt * 4 + nt) * 32 + lane) * 16);
        a2[nt] = WMMA_BF16(A, B2, a2[nt]);
      }
    }
    float s[8] = {0.f, 0.f, 0.f, 0.f, 0.f, 0.f, 0.f, 0.f};
#pragma unroll
    for (int nt = 0; nt < 4; ++nt)
#pragma unroll
      for (int j = 0; j < 8; ++j) {
        float x1 = a1[nt][j] + cb1[nt];
        float val = 1.f / (1.f + __expf(-x1)) + a2[nt][j] + cb2[nt];
        a1[nt][j] = val;
        s[j] += val * val;
      }
#pragma unroll
    for (int j = 0; j < 8; ++j) {
      float t = s[j];
      t += __shfl_xor(t, 1, 32);
      t += __shfl_xor(t, 2, 32);
      t += __shfl_xor(t, 4, 32);
      t += __shfl_xor(t, 8, 32);
      s[j] = t;
    }
#pragma unroll
    for (int j = 0; j < 8; ++j) {
      int row = j + hlf * 8;
      long gv = gvb + row;
      const int4* cvp = (const int4*)(clause_valid + gv * CN);
      int4 c0 = cvp[0], c1 = cvp[1];
      int has = (c0.x | c0.y | c0.z | c0.w | c1.x | c1.y | c1.z | c1.w) > 0;
      float inv = 1.f / fmaxf(sqrtf(s[j]), 1e-12f);
#pragma unroll
      for (int nt = 0; nt < 4; ++nt) {
        int col = nt * 16 + l16;
        float nv = a1[nt][j] * inv;
        float ov = Vin[gv * EN + col];
        Vout[gv * EN + col] = has ? nv : ov;
      }
    }
  }
}

extern "C" void kernel_launch(void* const* d_in, const int* in_sizes, int n_in,
                              void* d_out, int out_size, void* d_ws, size_t ws_size,
                              hipStream_t stream) {
  (void)in_sizes; (void)n_in; (void)out_size; (void)ws_size;
  const int*   clause_lits  = (const int*)d_in[0];
  const int*   lit_neg      = (const int*)d_in[1];
  const int*   lit_valid    = (const int*)d_in[2];
  const int*   clause_valid = (const int*)d_in[3];
  const float* emb_table    = (const float*)d_in[4];
  const float* false_emb    = (const float*)d_in[5];
  const float* tseitin_emb  = (const float*)d_in[6];
  const float* W_neg        = (const float*)d_in[7];
  const float* b_neg        = (const float*)d_in[8];
  const float* vc_W1        = (const float*)d_in[9];
  const float* vc_b1        = (const float*)d_in[10];
  const float* vc_W2        = (const float*)d_in[11];
  const float* vc_b2        = (const float*)d_in[12];
  const float* cc_W1        = (const float*)d_in[13];
  const float* cc_b1        = (const float*)d_in[14];
  const float* cc_W2        = (const float*)d_in[15];
  const float* cc_b2        = (const float*)d_in[16];

  unsigned short* frags = (unsigned short*)d_ws;                 // 237,568 B
  float* V0 = (float*)((char*)d_ws + 262144);                    // 5,120,000 B
  float* out = (float*)d_out;

  prep_weights_kernel<<<(NFRAG_TOT + 255) / 256, 256, 0, stream>>>(
      W_neg, vc_W1, vc_W2, cc_W1, cc_W2, frags);
  init_vars_kernel<<<VN, 32, 0, stream>>>(emb_table, tseitin_emb, V0);

  // iteration 1: V0 -> d_out ; iteration 2: d_out -> V0 ; copy back
  iter_kernel<<<VN / 16, 32, 0, stream>>>(V0, out, clause_lits, lit_neg, lit_valid,
                                          clause_valid, false_emb, b_neg, vc_b1, vc_b2,
                                          cc_b1, cc_b2, frags);
  iter_kernel<<<VN / 16, 32, 0, stream>>>(out, V0, clause_lits, lit_neg, lit_valid,
                                          clause_valid, false_emb, b_neg, vc_b1, vc_b2,
                                          cc_b1, cc_b2, frags);
  hipMemcpyAsync(d_out, V0, (size_t)VN * EN * sizeof(float),
                 hipMemcpyDeviceToDevice, stream);
}
